// CTR_59828894433719
// MI455X (gfx1250) — matrix operand order repaired and checked by reading, not compiled, not run on hardware
//
#include <hip/hip_runtime.h>
#include <hip/hip_bf16.h>

typedef __attribute__((ext_vector_type(16))) _Float16 v16h;
typedef __attribute__((ext_vector_type(8)))  _Float16 v8h;
typedef __attribute__((ext_vector_type(8)))  float    v8f;
typedef __attribute__((ext_vector_type(4)))  unsigned int u32x4;
typedef __attribute__((ext_vector_type(8)))  int      i32x8;
typedef __attribute__((ext_vector_type(4)))  int      i32x4;

#define B_    4
#define CIN   64
#define H_    256
#define W_    256
#define NPIX  65536
#define HID   128
#define KVC   256
#define HEAD  4

// ---- WMMA fragment index helpers (CDNA5 ISA 7.12.2, wave32, 16-bit A/B) ----
__device__ __forceinline__ int ka_idx(int i, int lane) {
    // A (16x32, MxK): m = lane&15 ; k below
    return i + ((lane >> 4) << 4);
}

// ---------------------------------------------------------------------------
// K1: bilinear 2x upsample (half-pixel centers), 128x128 -> 256x256 per plane
// ---------------------------------------------------------------------------
__global__ void upsample2x_kernel(const float* __restrict__ in, float* __restrict__ out) {
    size_t t = (size_t)blockIdx.x * blockDim.x + threadIdx.x;
    int pix = (int)(t & (NPIX - 1));
    size_t p = t >> 16;                 // plane index
    int x = pix & 255, y = pix >> 8;
    float sx = x * 0.5f - 0.25f;
    float sy = y * 0.5f - 0.25f;
    float x0f = floorf(sx), y0f = floorf(sy);
    float wx = sx - x0f, wy = sy - y0f;
    int x0 = (int)x0f, y0 = (int)y0f;
    int x0c = min(max(x0, 0), 127), x1c = min(max(x0 + 1, 0), 127);
    int y0c = min(max(y0, 0), 127), y1c = min(max(y0 + 1, 0), 127);
    const float* ip = in + p * (size_t)(128 * 128);
    float v00 = ip[y0c * 128 + x0c], v01 = ip[y0c * 128 + x1c];
    float v10 = ip[y1c * 128 + x0c], v11 = ip[y1c * 128 + x1c];
    float top = v00 + wx * (v01 - v00);
    float bot = v10 + wx * (v11 - v10);
    out[t] = top + wy * (bot - top);
}

// ---------------------------------------------------------------------------
// K2: backward warp of pre_h by flow_r (bilinear gather, OOB corners zeroed)
// ---------------------------------------------------------------------------
__global__ void warp_kernel(const float* __restrict__ pre_h, const float* __restrict__ flow_r,
                            _Float16* __restrict__ warped) {
    size_t t = (size_t)blockIdx.x * blockDim.x + threadIdx.x;
    int pix = (int)(t & (NPIX - 1));
    int b = (int)(t >> 16);
    float fx = flow_r[((size_t)b * 2 + 0) * NPIX + pix];
    float fy = flow_r[((size_t)b * 2 + 1) * NPIX + pix];
    float px = (float)(pix & 255) + fx;
    float py = (float)(pix >> 8)  + fy;
    float x0f = floorf(px), y0f = floorf(py);
    int x0 = (int)x0f, y0 = (int)y0f;
    float wx1 = px - x0f, wx0 = 1.0f - wx1;
    float wy1 = py - y0f, wy0 = 1.0f - wy1;
    bool vx0 = (x0 >= 0) && (x0 <= 255);
    bool vx1 = (x0 + 1 >= 0) && (x0 + 1 <= 255);
    bool vy0 = (y0 >= 0) && (y0 <= 255);
    bool vy1 = (y0 + 1 >= 0) && (y0 + 1 <= 255);
    float wa = (vx0 && vy0) ? wx0 * wy0 : 0.0f;
    float wb = (vx1 && vy0) ? wx1 * wy0 : 0.0f;
    float wc = (vx0 && vy1) ? wx0 * wy1 : 0.0f;
    float wd = (vx1 && vy1) ? wx1 * wy1 : 0.0f;
    int xc0 = min(max(x0, 0), 255), xc1 = min(max(x0 + 1, 0), 255);
    int yc0 = min(max(y0, 0), 255), yc1 = min(max(y0 + 1, 0), 255);
    int ia = yc0 * 256 + xc0, ib = yc0 * 256 + xc1;
    int ic = yc1 * 256 + xc0, id = yc1 * 256 + xc1;
    for (int c = 0; c < CIN; ++c) {
        const float* base = pre_h + ((size_t)b * CIN + c) * NPIX;
        float g = wa * base[ia] + wb * base[ib] + wc * base[ic] + wd * base[id];
        warped[((size_t)b * CIN + c) * NPIX + pix] = (_Float16)g;
    }
}

// ---------------------------------------------------------------------------
// K3/K4a: 1x1 conv (K=64) as GEMM via WMMA with LDS-staged transposed input.
// block = 128 (4 waves). Block tile: 64 pixels x 64 in-ch; each wave owns one
// 16-out-ch M-tile and sweeps 4 N-subtiles (A fragments kept in registers).
// grid.x = B*(NPIX/64), grid.y = M/64.
// ---------------------------------------------------------------------------
template <typename Tin>
__global__ void conv1x1_wmma(const Tin* __restrict__ in, const float* __restrict__ Wt,
                             const float* __restrict__ bias, _Float16* __restrict__ out, int M) {
    __shared__ _Float16 ldsT[64 * 72];          // 9216 B
    const int lane = threadIdx.x & 31;
    const int wave = threadIdx.x >> 5;
    const int tid = threadIdx.x;
    const int ntiles = NPIX / 64;
    int b = blockIdx.x / ntiles;
    size_t pix0 = (size_t)(blockIdx.x % ntiles) * 64;
    int m0 = (blockIdx.y * 4 + wave) * 16;

    // cooperative stage + transpose: global [k][n] (coalesced in n) -> ldsT[n][k]
#pragma unroll 4
    for (int it = 0; it < 32; ++it) {
        int flat = it * 128 + tid;              // 0..4095
        int k = flat >> 6, n = flat & 63;
        ldsT[n * 72 + k] = (_Float16)(float)in[((size_t)b * CIN + k) * NPIX + pix0 + n];
    }
    __syncthreads();

    // A fragments (weights) for the two K-steps, held in registers
    const int mrow = m0 + (lane & 15);
    v16h a0, a1;
#pragma unroll
    for (int i = 0; i < 16; ++i) {
        int ka = ka_idx(i, lane);
        a0[i] = (_Float16)Wt[(size_t)mrow * CIN + ka];
        a1[i] = (_Float16)Wt[(size_t)mrow * CIN + 32 + ka];
    }

    v8f acc[4] = {};
#pragma unroll
    for (int ns = 0; ns < 4; ++ns) {
        const _Float16* src = &ldsT[(ns * 16 + (lane & 15)) * 72 + ((lane >> 4) << 4)];
        v16h b0, b1;
#pragma unroll
        for (int i = 0; i < 16; ++i) { b0[i] = src[i]; b1[i] = src[32 + i]; }
        acc[ns] = __builtin_amdgcn_wmma_f32_16x16x32_f16(false, a0, false, b0, (short)0,
                                                         acc[ns], false, false);
        acc[ns] = __builtin_amdgcn_wmma_f32_16x16x32_f16(false, a1, false, b1, (short)0,
                                                         acc[ns], false, false);
    }

#pragma unroll
    for (int ns = 0; ns < 4; ++ns) {
#pragma unroll
        for (int r = 0; r < 8; ++r) {
            int m = m0 + r + ((lane >> 4) << 3);
            out[((size_t)b * M + m) * NPIX + pix0 + ns * 16 + (lane & 15)] =
                (_Float16)(acc[ns][r] + bias[m]);
        }
    }
}

// ---------------------------------------------------------------------------
// K3b/K4b: depthwise 3x3 conv via the Tensor Data Mover.
// Block = 256 threads owns one (b,c) plane x 8-row band. TDM DMAs the 10-row
// halo band (clamped) into LDS; wave 0 issues tensor_load_to_lds and waits on
// TENSORcnt; out-of-image halo rows are zero-filled into disjoint LDS rows.
// grid.x = B*C*32.
// ---------------------------------------------------------------------------
__global__ void dw3x3_tdm_kernel(const _Float16* __restrict__ in, const float* __restrict__ Wt,
                                 const float* __restrict__ bias, _Float16* __restrict__ out,
                                 int C) {
    __shared__ _Float16 tile[10 * 256];         // 5120 B ; row 0 == image row y0-1
    size_t bc = blockIdx.x >> 5;                // b*C + c
    int band = blockIdx.x & 31;
    int y0 = band * 8;
    int c = (int)(bc % C);
    int tid = threadIdx.x;

    int ys = max(y0 - 1, 0);
    int ye = min(y0 + 9, 256);
    int nrows = ye - ys;                        // 9 or 10
    int rs = ys - (y0 - 1);                     // first LDS row the TDM writes

    // zero-fill halo rows outside the image (disjoint from TDM target rows)
    if (y0 == 0)   tile[tid] = (_Float16)0.0f;            // row 0
    if (y0 == 248) tile[9 * 256 + tid] = (_Float16)0.0f;  // row 9

    if (tid < 32) {
        // Tensor DMA descriptor (CDNA5 ISA 8.3/8.4), 2D tile: 256 x nrows, 2B elems
        unsigned long long ga =
            (unsigned long long)(uintptr_t)(in + bc * NPIX + (size_t)ys * 256);
        unsigned int lds_addr =
            (unsigned int)(uintptr_t)(&tile[0]) + (unsigned int)(rs * 512);
        u32x4 g0;
        g0[0] = 1u;                                        // count=1, user mode
        g0[1] = lds_addr;                                  // lds_addr [63:32]
        g0[2] = (unsigned int)(ga & 0xffffffffu);          // global_addr low
        g0[3] = (unsigned int)((ga >> 32) & 0x1ffffffu) | (2u << 30);  // ga hi | type=2
        i32x8 g1;
        g1[0] = 0x00010000;                                // data_size = 2 bytes
        g1[1] = (int)(256u << 16);                         // tensor_dim0 = 256
        g1[2] = (int)((unsigned)nrows << 16);              // tensor_dim1 = nrows
        g1[3] = (int)(256u << 16);                         // tile_dim0 = 256
        g1[4] = nrows;                                     // tile_dim1 = nrows
        g1[5] = 256;                                       // tensor_dim0_stride = 256
        g1[6] = 0;
        g1[7] = 0;
        i32x4 gz4 = {0, 0, 0, 0};
        i32x8 gz8 = {0, 0, 0, 0, 0, 0, 0, 0};
        __builtin_amdgcn_tensor_load_to_lds(g0, g1, gz4, gz4, gz8, 0);
        __builtin_amdgcn_s_wait_tensorcnt(0);
    }
    asm volatile("" ::: "memory");
    __syncthreads();

    const float* wkp = Wt + (size_t)c * 9;
    float wk[9];
#pragma unroll
    for (int i = 0; i < 9; ++i) wk[i] = wkp[i];
    float bs = bias[c];

    int x = tid;                                 // 0..255
#pragma unroll
    for (int it = 0; it < 8; ++it) {
        float s = bs;
#pragma unroll
        for (int dy = 0; dy < 3; ++dy) {
            const _Float16* row = &tile[(it + dy) * 256];
            if (x > 0)   s += (float)row[x - 1] * wk[dy * 3 + 0];
                         s += (float)row[x]     * wk[dy * 3 + 1];
            if (x < 255) s += (float)row[x + 1] * wk[dy * 3 + 2];
        }
        out[bc * NPIX + (size_t)(y0 + it) * 256 + x] = (_Float16)s;
    }
}

// ---------------------------------------------------------------------------
// K5a: per-channel L2 norm over n=65536 (one 256-thread block per row)
// ---------------------------------------------------------------------------
__global__ void row_norm_kernel(const _Float16* __restrict__ in, float* __restrict__ outn,
                                int CH, int CHS) {
    __shared__ float red[256];
    int r = blockIdx.x;
    int b = r / CH, c = r % CH;
    const v8h* basev = (const v8h*)(in + ((size_t)b * CHS + c) * NPIX);
    float s = 0.0f;
    for (int i = threadIdx.x; i < NPIX / 8; i += 256) {
        v8h v = basev[i];
#pragma unroll
        for (int j = 0; j < 8; ++j) {
            float f = (float)v[j];
            s += f * f;
        }
    }
    red[threadIdx.x] = s;
    __syncthreads();
    for (int st = 128; st > 0; st >>= 1) {
        if ((int)threadIdx.x < st) red[threadIdx.x] += red[threadIdx.x + st];
        __syncthreads();
    }
    if (threadIdx.x == 0) outn[r] = fmaxf(sqrtf(red[0]), 1e-12f);
}

// ---------------------------------------------------------------------------
// K5b: Gram matrix D[c,e] = sum_n q1[c,n]*k[e,n] per (b,head) via WMMA.
// ---------------------------------------------------------------------------
__global__ void gram_wmma(const _Float16* __restrict__ q, const _Float16* __restrict__ kv,
                          float* __restrict__ Dpart) {
    const int lane = threadIdx.x & 31;
    const int wave = threadIdx.x >> 5;
    int bh = blockIdx.x & 15;          // b*HEAD + hd
    int chunk = blockIdx.x >> 4;
    int b = bh >> 2, hd = bh & 3;
    size_t n0 = ((size_t)chunk * 4 + wave) * 2048;
    const _Float16* qrow = q + ((size_t)b * HID + hd * 16 + (lane & 15)) * NPIX + n0;
    const _Float16* krow0 = kv + ((size_t)b * KVC + hd * 32 + (lane & 15)) * NPIX + n0;
    const _Float16* krow1 = krow0 + (size_t)16 * NPIX;
    v8f acc0 = {}, acc1 = {};
    for (int nn = 0; nn < 2048; nn += 32) {
        v16h a, b0, b1;
#pragma unroll
        for (int i = 0; i < 16; ++i) a[i] = qrow[nn + ka_idx(i, lane)];
#pragma unroll
        for (int i = 0; i < 16; ++i) {
            int k = nn + ((lane >> 4) << 4) + i;
            b0[i] = krow0[k];
            b1[i] = krow1[k];
        }
        acc0 = __builtin_amdgcn_wmma_f32_16x16x32_f16(false, a, false, b0, (short)0, acc0,
                                                      false, false);
        acc1 = __builtin_amdgcn_wmma_f32_16x16x32_f16(false, a, false, b1, (short)0, acc1,
                                                      false, false);
    }
    int p = chunk * 4 + wave;          // 0..31
    float* dst = Dpart + ((size_t)p * 16 + bh) * (16 * 32);
#pragma unroll
    for (int r = 0; r < 8; ++r) {
        int m = r + ((lane >> 4) << 3);
        int e = lane & 15;
        dst[m * 32 + e]      = acc0[r];
        dst[m * 32 + e + 16] = acc1[r];
    }
}

// ---------------------------------------------------------------------------
// K6a: reduce partials (fixed order), apply norms, softmax over e.
// ---------------------------------------------------------------------------
__global__ void attn_softmax_kernel(const float* __restrict__ Dpart, const float* __restrict__ nq,
                                    const float* __restrict__ nk, float* __restrict__ attn) {
    int bh = blockIdx.x;
    int c = threadIdx.x;               // 0..15
    int b = bh >> 2, hd = bh & 3;
    float lo[32];
    float mx = -1e30f;
    for (int e = 0; e < 32; ++e) {
        float s = 0.0f;
        for (int p = 0; p < 32; ++p)
            s += Dpart[((size_t)p * 16 + bh) * (16 * 32) + c * 32 + e];
        float d = nq[b * 64 + hd * 16 + c] * nk[b * 128 + hd * 32 + e];
        float l = s / d + 1e-7f;
        lo[e] = l;
        mx = fmaxf(mx, l);
    }
    float sum = 0.0f;
    for (int e = 0; e < 32; ++e) { lo[e] = expf(lo[e] - mx); sum += lo[e]; }
    float inv = 1.0f / sum;
    for (int e = 0; e < 32; ++e)
        attn[((size_t)bh * 16 + c) * 32 + e] = lo[e] * inv;
}

// ---------------------------------------------------------------------------
// K6b: fold attention into final projection weights (64 x 224, f16, per b).
// ---------------------------------------------------------------------------
__global__ void build_wcomb(const float* __restrict__ attn, const float* __restrict__ wp,
                            _Float16* __restrict__ wcomb) {
    int t = blockIdx.x * blockDim.x + threadIdx.x;
    if (t >= B_ * 64 * 224) return;
    int j = t % 224;
    int o = (t / 224) % 64;
    int b = t / (224 * 64);
    float v = 0.0f;
    if (j < 128) {
        int hd = j >> 5, e = j & 31;
        for (int c = 0; c < 16; ++c)
            v += wp[o * 129 + hd * 16 + c] * attn[(((size_t)b * 4 + hd) * 16 + c) * 32 + e];
    } else if (j < 192) {
        v = wp[o * 129 + 64 + (j - 128)];
    } else if (j == 192) {
        v = wp[o * 129 + 128];
    }
    wcomb[t] = (_Float16)v;
}

// ---------------------------------------------------------------------------
// K7: final fused GEMM: out = Wcomb_b[64x224] . cat[v(128)|q2(64)|depth|pad].
// ---------------------------------------------------------------------------
__global__ void final_wmma(const _Float16* __restrict__ qbuf, const _Float16* __restrict__ kvbuf,
                           const float* __restrict__ depth_r, const _Float16* __restrict__ wcomb,
                           const float* __restrict__ bp, float* __restrict__ out) {
    __shared__ _Float16 ldsT[64 * 232];         // 29696 B
    const int lane = threadIdx.x & 31;
    const int wave = threadIdx.x >> 5;
    const int tid = threadIdx.x;
    const int ntiles = NPIX / 64;
    int b = blockIdx.x / ntiles;
    size_t pix0 = (size_t)(blockIdx.x % ntiles) * 64;
    int m0 = wave * 16;

    // stage concatenated K rows: [0,128) v ; [128,192) q2 ; 192 depth ; pad 0
#pragma unroll 4
    for (int it = 0; it < 112; ++it) {
        int flat = it * 128 + tid;              // 0..14335
        int j = flat >> 6, n = flat & 63;
        float v;
        if (j < 128)
            v = (float)kvbuf[((size_t)b * KVC + 128 + j) * NPIX + pix0 + n];
        else if (j < 192)
            v = (float)qbuf[((size_t)b * HID + 64 + (j - 128)) * NPIX + pix0 + n];
        else if (j == 192)
            v = depth_r[(size_t)b * NPIX + pix0 + n];
        else
            v = 0.0f;
        ldsT[n * 232 + j] = (_Float16)v;
    }
    __syncthreads();

    const _Float16* wrow = wcomb + ((size_t)b * 64 + m0 + (lane & 15)) * 224;
    v8f acc[4] = {};
    for (int kk = 0; kk < 224; kk += 32) {
        v16h a;
#pragma unroll
        for (int i = 0; i < 16; ++i) a[i] = wrow[kk + ka_idx(i, lane)];
#pragma unroll
        for (int ns = 0; ns < 4; ++ns) {
            const _Float16* src =
                &ldsT[(ns * 16 + (lane & 15)) * 232 + kk + ((lane >> 4) << 4)];
            v16h bf;
#pragma unroll
            for (int i = 0; i < 16; ++i) bf[i] = src[i];
            acc[ns] = __builtin_amdgcn_wmma_f32_16x16x32_f16(false, a, false, bf, (short)0,
                                                             acc[ns], false, false);
        }
    }

#pragma unroll
    for (int ns = 0; ns < 4; ++ns) {
#pragma unroll
        for (int r = 0; r < 8; ++r) {
            int m = m0 + r + ((lane >> 4) << 3);
            out[((size_t)b * 64 + m) * NPIX + pix0 + ns * 16 + (lane & 15)] =
                acc[ns][r] + bp[m];
        }
    }
}

// ---------------------------------------------------------------------------
extern "C" void kernel_launch(void* const* d_in, const int* in_sizes, int n_in,
                              void* d_out, int out_size, void* d_ws, size_t ws_size,
                              hipStream_t stream) {
    const float* x     = (const float*)d_in[0];
    const float* depth = (const float*)d_in[1];
    const float* pre_h = (const float*)d_in[2];
    const float* flow  = (const float*)d_in[3];
    const float* wq1   = (const float*)d_in[4];
    const float* bq1   = (const float*)d_in[5];
    const float* wq2   = (const float*)d_in[6];
    const float* bq2   = (const float*)d_in[7];
    const float* wkv1  = (const float*)d_in[8];
    const float* bkv1  = (const float*)d_in[9];
    const float* wkv2  = (const float*)d_in[10];
    const float* bkv2  = (const float*)d_in[11];
    const float* wp    = (const float*)d_in[12];
    const float* bp    = (const float*)d_in[13];
    float* out = (float*)d_out;

    // ---- workspace layout (256B aligned) ----
    char* w = (char*)d_ws;
    size_t off = 0;
    auto take = [&](size_t bytes) -> char* {
        char* p = w + off;
        off = (off + bytes + 255) & ~(size_t)255;
        return p;
    };
    float*    flow_r  = (float*)take((size_t)B_ * 2 * NPIX * 4);       //  2 MB
    float*    depth_r = (float*)take((size_t)B_ * NPIX * 4);           //  1 MB
    _Float16* warped  = (_Float16*)take((size_t)B_ * CIN * NPIX * 2);  // 33.5 MB
    _Float16* pre     = (_Float16*)take((size_t)B_ * KVC * NPIX * 2);  // 134 MB (shared q_pre/kv_pre)
    _Float16* qbuf    = (_Float16*)take((size_t)B_ * HID * NPIX * 2);  // 67 MB
    _Float16* kvbuf   = (_Float16*)take((size_t)B_ * KVC * NPIX * 2);  // 134 MB
    float*    Dpart   = (float*)take((size_t)32 * 16 * 16 * 32 * 4);   //  1 MB
    float*    nq      = (float*)take((size_t)B_ * 64 * 4);
    float*    nk      = (float*)take((size_t)B_ * 128 * 4);
    float*    attn    = (float*)take((size_t)16 * 16 * 32 * 4);
    _Float16* wcomb   = (_Float16*)take((size_t)B_ * 64 * 224 * 2);
    (void)ws_size; (void)in_sizes; (void)n_in; (void)out_size;

    const int ntilesN64 = B_ * (NPIX / 64);   // 4096

    // K1: upsample flow (8 planes) and depth (4 planes)
    upsample2x_kernel<<<(B_ * 2 * NPIX) / 256, 256, 0, stream>>>(flow, flow_r);
    upsample2x_kernel<<<(B_ * 1 * NPIX) / 256, 256, 0, stream>>>(depth, depth_r);

    // K2: warp pre_h by flow_r -> warped (f16)
    warp_kernel<<<(B_ * NPIX) / 256, 256, 0, stream>>>(pre_h, flow_r, warped);

    // K3: q path: 1x1 conv (WMMA, LDS-staged) then depthwise 3x3 (TDM-staged)
    conv1x1_wmma<float><<<dim3(ntilesN64, HID / 64), 128, 0, stream>>>(x, wq1, bq1, pre, HID);
    dw3x3_tdm_kernel<<<B_ * HID * 32, 256, 0, stream>>>(pre, wq2, bq2, qbuf, HID);

    // K4: kv path: 1x1 conv (WMMA, LDS-staged) then depthwise 3x3 (TDM-staged)
    conv1x1_wmma<_Float16><<<dim3(ntilesN64, KVC / 64), 128, 0, stream>>>(warped, wkv1, bkv1,
                                                                         pre, KVC);
    dw3x3_tdm_kernel<<<B_ * KVC * 32, 256, 0, stream>>>(pre, wkv2, bkv2, kvbuf, KVC);

    // K5: per-channel L2 norms + Gram via WMMA (deterministic partials)
    row_norm_kernel<<<B_ * 64, 256, 0, stream>>>(qbuf, nq, 64, HID);
    row_norm_kernel<<<B_ * 128, 256, 0, stream>>>(kvbuf, nk, 128, KVC);
    gram_wmma<<<8 * 16, 128, 0, stream>>>(qbuf, kvbuf, Dpart);

    // K6: softmax + fold attention into final projection weights
    attn_softmax_kernel<<<16, 16, 0, stream>>>(Dpart, nq, nk, attn);
    build_wcomb<<<(B_ * 64 * 224 + 255) / 256, 256, 0, stream>>>(attn, wp, wcomb);

    // K7: fused (attn.v | q2 | depth) -> 1x1 projection, WMMA GEMM -> d_out
    final_wmma<<<dim3(ntilesN64, 1), 128, 0, stream>>>(qbuf, kvbuf, depth_r, wcomb, bp, out);
}
